// VGAEModel_76733885710552
// MI455X (gfx1250) — matrix operand order repaired and checked by reading, not compiled, not run on hardware
//
#include <hip/hip_runtime.h>
#include <hip/hip_bf16.h>

typedef __attribute__((ext_vector_type(16))) _Float16 v16h;
typedef __attribute__((ext_vector_type(8)))  _Float16 v8h;
typedef __attribute__((ext_vector_type(8)))  float    v8f;

#define CDIV(a, b) (((a) + (b) - 1) / (b))

// ---------------------------------------------------------------------------
// WMMA fragment loaders (wave32, v_wmma_f32_16x16x32_f16 layouts per CDNA5 ISA)
// ---------------------------------------------------------------------------

// A: 16x32 f16 tile, row-major source with row stride `ld` (elements).
// Lane L (0..15): row M=L, K = 0..7 (v0..3) and 16..23 (v4..7).
// Lane L (16..31): row M=L-16, K = 8..15 and 24..31.
// Two contiguous 16B loads per lane.
__device__ __forceinline__ v16h load_a_frag(const _Float16* __restrict__ base, int ld) {
  int lane  = threadIdx.x & 31;
  int m     = lane & 15;
  int khalf = (lane >> 4) << 3;  // 0 or 8
  const v8h* p0 = (const v8h*)(base + (size_t)m * ld + khalf);
  const v8h* p1 = (const v8h*)(base + (size_t)m * ld + khalf + 16);
  v8h lo = *p0, hi = *p1;
  v16h a;
#pragma unroll
  for (int i = 0; i < 8; ++i) { a[i] = lo[i]; a[i + 8] = hi[i]; }
  return a;
}

// B fragment where B[k][n] = T[n0+n][k] and T is row-major with stride ld
// (T = W^T for the layer GEMMs, T = Z for the z z^T decode).
// trow0 points at T[n0][k0]. ISA layout: lane L holds N = L&15,
// elem e<8: K = ((L>>4)<<3) + e; e>=8: K = 16 + ((L>>4)<<3) + (e-8),
// i.e. two contiguous 16B loads per lane.
__device__ __forceinline__ v16h load_bt_frag(const _Float16* __restrict__ trow0, int ld) {
  int lane = threadIdx.x & 31;
  int n    = lane & 15;
  int kofs = (lane >> 4) << 3;  // 0 or 8
  const v8h* p = (const v8h*)(trow0 + (size_t)n * ld + kofs);
  v8h lo = p[0], hi = p[2];
  v16h b;
#pragma unroll
  for (int i = 0; i < 8; ++i) { b[i] = lo[i]; b[i + 8] = hi[i]; }
  return b;
}

// ---------------------------------------------------------------------------
// GEMM: C[M,Nc] = A[M,K] (f16 row-major) @ BT[Nc,K]^T (f16 row-major), f32 out.
// B is supplied pre-transposed so every operand load is a B128 load.
// One wave per 16x16 output tile.
// ---------------------------------------------------------------------------
__global__ void k_gemm_f16(const _Float16* __restrict__ A, const _Float16* __restrict__ BT,
                           float* __restrict__ C, int M, int K, int Nc) {
  int wid    = (int)((blockIdx.x * blockDim.x + threadIdx.x) >> 5);
  int lane   = threadIdx.x & 31;
  int tilesN = Nc >> 4;
  int ti     = wid / tilesN;
  int tj     = wid - ti * tilesN;
  if (ti >= (M >> 4)) return;  // uniform per wave -> EXEC stays all-1s

  v8f acc = {};
  for (int k0 = 0; k0 < K; k0 += 32) {
    v16h a = load_a_frag(A + (size_t)(ti << 4) * K + k0, K);
    v16h b = load_bt_frag(BT + (size_t)(tj << 4) * K + k0, K);
    acc = __builtin_amdgcn_wmma_f32_16x16x32_f16(false, a, false, b, (short)0, acc,
                                                 false, false);
  }
  int ml = (lane >> 4) << 3;  // 0 or 8
  int nl = lane & 15;
  float* crow = C + (size_t)((ti << 4) + ml) * Nc + (tj << 4) + nl;
#pragma unroll
  for (int r = 0; r < 8; ++r) crow[(size_t)r * Nc] = acc[r];
}

// ---------------------------------------------------------------------------
// Decode: out = sigmoid(Z @ Z^T), Z = [n, 64] f16. One wave per row of 4 16x16 tiles,
// A fragments reused across 4 consecutive j-tiles. Store-BW bound on MI455X.
// ---------------------------------------------------------------------------
__global__ void k_decode(const _Float16* __restrict__ Z, float* __restrict__ out, int n) {
  const int ldz = 64;
  int wid   = (int)((blockIdx.x * blockDim.x + threadIdx.x) >> 5);
  int lane  = threadIdx.x & 31;
  int tiles = n >> 4;
  int jg    = CDIV(tiles, 4);
  int ti    = wid / jg;
  int tg    = wid - ti * jg;
  if (ti >= tiles) return;

  v16h a0 = load_a_frag(Z + (size_t)(ti << 4) * ldz, ldz);
  v16h a1 = load_a_frag(Z + (size_t)(ti << 4) * ldz + 32, ldz);
  int ml = (lane >> 4) << 3;
  int nl = lane & 15;

#pragma unroll
  for (int jj = 0; jj < 4; ++jj) {
    int tj = tg * 4 + jj;
    if (tj >= tiles) break;  // uniform per wave
    const _Float16* zr = Z + (size_t)(tj << 4) * ldz;
    if (jj < 3 && tj + 1 < tiles)
      __builtin_prefetch(Z + (size_t)((tj + 1) << 4) * ldz, 0, 1);
    v16h b0 = load_bt_frag(zr, ldz);
    v16h b1 = load_bt_frag(zr + 32, ldz);
    v8f acc = {};
    acc = __builtin_amdgcn_wmma_f32_16x16x32_f16(false, a0, false, b0, (short)0, acc,
                                                 false, false);
    acc = __builtin_amdgcn_wmma_f32_16x16x32_f16(false, a1, false, b1, (short)0, acc,
                                                 false, false);
    float* crow = out + (size_t)((ti << 4) + ml) * n + (tj << 4) + nl;
#pragma unroll
    for (int r = 0; r < 8; ++r)
      crow[(size_t)r * n] = 1.0f / (1.0f + __expf(-acc[r]));
  }
}

// ---------------------------------------------------------------------------
// Graph norm + aggregation helpers
// ---------------------------------------------------------------------------
__global__ void k_fill_ones(float* d, int n) {
  int i = blockIdx.x * blockDim.x + threadIdx.x;
  if (i < n) d[i] = 1.0f;  // self-loop contribution to degree
}

__global__ void k_deg_edges(const int* __restrict__ col, const float* __restrict__ w,
                            float* deg, int E) {
  int i = blockIdx.x * blockDim.x + threadIdx.x;
  if (i < E) atomicAdd(&deg[col[i]], w[i]);
}

__global__ void k_dinv(const float* __restrict__ deg, float* dinv, int n) {
  int i = blockIdx.x * blockDim.x + threadIdx.x;
  if (i < n) {
    float d = deg[i];
    dinv[i] = d > 0.0f ? rsqrtf(d) : 0.0f;
  }
}

__global__ void k_enorm(const int* __restrict__ row, const int* __restrict__ col,
                        const float* __restrict__ w, const float* __restrict__ dinv,
                        float* enorm, int E) {
  int i = blockIdx.x * blockDim.x + threadIdx.x;
  if (i < E) enorm[i] = dinv[row[i]] * w[i] * dinv[col[i]];
}

__global__ void k_cvt_h(const float* __restrict__ x, _Float16* __restrict__ y, int n) {
  int i = blockIdx.x * blockDim.x + threadIdx.x;
  if (i < n) y[i] = (_Float16)x[i];
}

// Convert f32 W[K,Nc] -> f16 WT[Nc,K] (row-major transpose). Coalesced read.
__global__ void k_cvt_h_tr(const float* __restrict__ w, _Float16* __restrict__ wt,
                           int K, int Nc) {
  int i = blockIdx.x * blockDim.x + threadIdx.x;
  if (i < K * Nc) {
    int r = i / Nc;
    int c = i - r * Nc;
    wt[(size_t)c * K + r] = (_Float16)w[i];
  }
}

// out[i,f] = bias[f] + dinv[i]^2 * hw[i,f]   (self-loop folded in)
template <int F>
__global__ void k_agg_init(const float* __restrict__ hw, const float* __restrict__ bias,
                           const float* __restrict__ dinv, float* __restrict__ out,
                           int total) {
  int i = blockIdx.x * blockDim.x + threadIdx.x;
  if (i < total) {
    int node = i / F;
    int f    = i - node * F;
    float di = dinv[node];
    out[i] = bias[f] + di * di * hw[i];
  }
}

// out[col[e], f] += enorm[e] * hw[row[e], f]
template <int F>
__global__ void k_agg_edges(const float* __restrict__ hw, const int* __restrict__ row,
                            const int* __restrict__ col, const float* __restrict__ enorm,
                            float* out, int E) {
  int i = blockIdx.x * blockDim.x + threadIdx.x;
  if (i >= E * F) return;
  int e = i / F;
  int f = i - e * F;
  atomicAdd(&out[(size_t)col[e] * F + f], enorm[e] * hw[(size_t)row[e] * F + f]);
}

__global__ void k_relu(float* h, int n) {
  int i = blockIdx.x * blockDim.x + threadIdx.x;
  if (i < n) h[i] = fmaxf(h[i], 0.0f);
}

// z = mean + noise * exp(log_std), emitted directly as f16 for the decode GEMM
__global__ void k_z(const float* __restrict__ mean, const float* __restrict__ lstd,
                    const float* __restrict__ noise, _Float16* __restrict__ zh, int n) {
  int i = blockIdx.x * blockDim.x + threadIdx.x;
  if (i < n) zh[i] = (_Float16)(mean[i] + noise[i] * __expf(lstd[i]));
}

// ---------------------------------------------------------------------------
extern "C" void kernel_launch(void* const* d_in, const int* in_sizes, int n_in,
                              void* d_out, int out_size, void* d_ws, size_t ws_size,
                              hipStream_t stream) {
  const int D = 256, H1 = 128, H2 = 64;
  const float* x     = (const float*)d_in[0];
  const int*   ei    = (const int*)d_in[1];
  const float* ew    = (const float*)d_in[2];
  const float* noise = (const float*)d_in[3];
  const float* W1    = (const float*)d_in[4];
  const float* b1    = (const float*)d_in[5];
  const float* W2    = (const float*)d_in[6];
  const float* b2    = (const float*)d_in[7];
  const float* W3    = (const float*)d_in[8];
  const float* b3    = (const float*)d_in[9];

  const int N = in_sizes[0] / D;   // 10000 (divisible by 16)
  const int E = in_sizes[2];       // 160000
  const int* row = ei;
  const int* col = ei + E;
  float* out = (float*)d_out;

  // Carve workspace (everything we read is written first; ws is poisoned 0xAA).
  char* p = (char*)d_ws;
  auto carveF = [&](size_t n) -> float* {
    float* r = (float*)p; p += ((n * 4 + 255) / 256) * 256; return r;
  };
  auto carveH = [&](size_t n) -> _Float16* {
    _Float16* r = (_Float16*)p; p += ((n * 2 + 255) / 256) * 256; return r;
  };
  float* deg   = carveF(N);
  float* dinv  = carveF(N);
  float* enorm = carveF(E);
  float* hw1   = carveF((size_t)N * H1);
  float* h     = carveF((size_t)N * H1);
  float* hw2   = carveF((size_t)N * H2);
  float* mean  = carveF((size_t)N * H2);
  float* hw3   = carveF((size_t)N * H2);
  float* lstd  = carveF((size_t)N * H2);
  _Float16* xh   = carveH((size_t)N * D);
  _Float16* w1t  = carveH((size_t)D * H1);   // W1^T: [H1, D]
  _Float16* w2t  = carveH((size_t)H1 * H2);  // W2^T: [H2, H1]
  _Float16* w3t  = carveH((size_t)H1 * H2);  // W3^T: [H2, H1]
  _Float16* hh   = carveH((size_t)N * H1);
  _Float16* zh   = carveH((size_t)N * H2);

  const int B = 256;  // 8 waves of 32 on CDNA5

  // GCN normalization
  k_fill_ones<<<CDIV(N, B), B, 0, stream>>>(deg, N);
  k_deg_edges<<<CDIV(E, B), B, 0, stream>>>(col, ew, deg, E);
  k_dinv<<<CDIV(N, B), B, 0, stream>>>(deg, dinv, N);
  k_enorm<<<CDIV(E, B), B, 0, stream>>>(row, col, ew, dinv, enorm, E);

  // f16 operand copies (weights transposed so B-fragment loads are B128)
  k_cvt_h<<<CDIV(N * D, B), B, 0, stream>>>(x, xh, N * D);
  k_cvt_h_tr<<<CDIV(D * H1, B), B, 0, stream>>>(W1, w1t, D, H1);
  k_cvt_h_tr<<<CDIV(H1 * H2, B), B, 0, stream>>>(W2, w2t, H1, H2);
  k_cvt_h_tr<<<CDIV(H1 * H2, B), B, 0, stream>>>(W3, w3t, H1, H2);

  // Layer 1: h = relu(aggregate(x @ W1) + b1)
  {
    int waves = (N / 16) * (H1 / 16);
    k_gemm_f16<<<CDIV(waves * 32, B), B, 0, stream>>>(xh, w1t, hw1, N, D, H1);
  }
  k_agg_init<128><<<CDIV(N * H1, B), B, 0, stream>>>(hw1, b1, dinv, h, N * H1);
  k_agg_edges<128><<<CDIV(E * 128, B), B, 0, stream>>>(hw1, row, col, enorm, h, E);
  k_relu<<<CDIV(N * H1, B), B, 0, stream>>>(h, N * H1);
  k_cvt_h<<<CDIV(N * H1, B), B, 0, stream>>>(h, hh, N * H1);

  // mean = aggregate(h @ W2) + b2
  {
    int waves = (N / 16) * (H2 / 16);
    k_gemm_f16<<<CDIV(waves * 32, B), B, 0, stream>>>(hh, w2t, hw2, N, H1, H2);
  }
  k_agg_init<64><<<CDIV(N * H2, B), B, 0, stream>>>(hw2, b2, dinv, mean, N * H2);
  k_agg_edges<64><<<CDIV(E * 64, B), B, 0, stream>>>(hw2, row, col, enorm, mean, E);

  // log_std = aggregate(h @ W3) + b3
  {
    int waves = (N / 16) * (H2 / 16);
    k_gemm_f16<<<CDIV(waves * 32, B), B, 0, stream>>>(hh, w3t, hw3, N, H1, H2);
  }
  k_agg_init<64><<<CDIV(N * H2, B), B, 0, stream>>>(hw3, b3, dinv, lstd, N * H2);
  k_agg_edges<64><<<CDIV(E * 64, B), B, 0, stream>>>(hw3, row, col, enorm, lstd, E);

  // z (f16) then decode sigmoid(z z^T) — dominant, store-bandwidth bound
  k_z<<<CDIV(N * H2, B), B, 0, stream>>>(mean, lstd, noise, zh, N * H2);
  {
    int tiles = N / 16;
    int jg = CDIV(tiles, 4);
    long long waves = (long long)tiles * jg;
    k_decode<<<(int)CDIV(waves * 32, (long long)B), B, 0, stream>>>(zh, out, N);
  }
}
